// PointNetSetAbstraction_34711925687087
// MI455X (gfx1250) — matrix operand order repaired and checked by reading, not compiled, not run on hardware
//
#include <hip/hip_runtime.h>
#include <hip/hip_bf16.h>

#define BB   16
#define NN   4096
#define CIN  64
#define NPT  1024
#define NSMP 32
#define BIGF 1e10f

typedef __attribute__((ext_vector_type(16))) _Float16 v16h;
typedef __attribute__((ext_vector_type(8)))  _Float16 v8h;
typedef __attribute__((ext_vector_type(8)))  float    v8f;

// ---------------------------------------------------------------------------
// Kernel 1: farthest point sampling (faithful to reference semantics).
// One block per batch. xyz staged in LDS; per-thread distance state in regs.
// ---------------------------------------------------------------------------
__global__ __launch_bounds__(256) void fps_kernel(const float* __restrict__ xyz,
                                                  const int* __restrict__ mask,
                                                  int* __restrict__ fps_idx) {
  __shared__ float sX[NN], sY[NN], sZ[NN];
  __shared__ float rv[256];
  __shared__ int   ri[256];
  __shared__ int   sfar;
  const int b = blockIdx.x;
  const int t = threadIdx.x;
  const float* bx = xyz + (size_t)b * NN * 3;

  for (int i = t; i < NN; i += 256) {
    sX[i] = bx[i * 3 + 0];
    sY[i] = bx[i * 3 + 1];
    sZ[i] = bx[i * 3 + 2];
  }

  // per-thread blocked ownership of 16 points: distance + mask bits in regs
  float pd[16];
  unsigned pm = 0u;
  float lv = -1.f; int li = 0;
  #pragma unroll
  for (int j = 0; j < 16; ++j) {
    const int i = t * 16 + j;
    const int m = mask[(size_t)b * NN + i];
    pd[j] = m ? 0.f : BIGF;                 // reference: where(mask, 0, BIG)
    if (m) pm |= (1u << j);
    const float v = m ? 1.f : 0.f;          // initial farthest = argmax(mask)
    if (v > lv) { lv = v; li = i; }
  }
  rv[t] = lv; ri[t] = li;
  __syncthreads();
  for (int s2 = 128; s2 > 0; s2 >>= 1) {
    if (t < s2) {
      if (rv[t + s2] > rv[t] || (rv[t + s2] == rv[t] && ri[t + s2] < ri[t])) {
        rv[t] = rv[t + s2]; ri[t] = ri[t + s2];
      }
    }
    __syncthreads();
  }
  if (t == 0) sfar = ri[0];
  __syncthreads();

  for (int it = 0; it < NPT; ++it) {
    const int far = sfar;
    if (t == 0) fps_idx[(size_t)b * NPT + it] = far;   // emit index at step start
    const float cx = sX[far], cy = sY[far], cz = sZ[far];
    lv = -1.f; li = 0;
    #pragma unroll
    for (int j = 0; j < 16; ++j) {
      const int i = t * 16 + j;
      const float dx = sX[i] - cx, dy = sY[i] - cy, dz = sZ[i] - cz;
      const float d = dx * dx + dy * dy + dz * dz;
      if ((d < pd[j]) && ((pm >> j) & 1u)) pd[j] = d;
      if (pd[j] > lv) { lv = pd[j]; li = i; }          // first-occurrence kept
    }
    __syncthreads();
    rv[t] = lv; ri[t] = li;
    __syncthreads();
    for (int s2 = 128; s2 > 0; s2 >>= 1) {
      if (t < s2) {
        if (rv[t + s2] > rv[t] || (rv[t + s2] == rv[t] && ri[t + s2] < ri[t])) {
          rv[t] = rv[t + s2]; ri[t] = ri[t + s2];
        }
      }
      __syncthreads();
    }
    if (t == 0) sfar = ri[0];
    __syncthreads();
  }
}

// ---------------------------------------------------------------------------
// Kernel 2: kNN (32 smallest distances) + new_xyz emission.
// One block per query; distance array in LDS; 32 masked argmin rounds.
// ---------------------------------------------------------------------------
__global__ __launch_bounds__(256) void knn_kernel(const float* __restrict__ xyz,
                                                  const int* __restrict__ mask,
                                                  const int* __restrict__ fps_idx,
                                                  int* __restrict__ knn_idx,
                                                  float* __restrict__ out_xyz) {
  __shared__ float sd[NN];
  __shared__ float rv[256];
  __shared__ int   ri[256];
  __shared__ int   sres[NSMP];
  __shared__ int   svalid;
  const int t = threadIdx.x;
  const int b = blockIdx.x >> 10;          // NPT == 1024
  const int s = blockIdx.x & (NPT - 1);
  const float* bx = xyz + (size_t)b * NN * 3;
  const int fi = fps_idx[(size_t)b * NPT + s];
  const float qx = bx[fi * 3 + 0], qy = bx[fi * 3 + 1], qz = bx[fi * 3 + 2];
  if (t < 3) out_xyz[((size_t)b * NPT + s) * 3 + t] = bx[fi * 3 + t];

  int lc = 0;
  for (int i = t; i < NN; i += 256) {
    const int m = mask[(size_t)b * NN + i];
    const float dx = bx[i * 3 + 0] - qx;
    const float dy = bx[i * 3 + 1] - qy;
    const float dz = bx[i * 3 + 2] - qz;
    const float d = dx * dx + dy * dy + dz * dz;
    sd[i] = m ? d : BIGF;
    lc += m ? 1 : 0;
  }
  ri[t] = lc;
  __syncthreads();
  for (int s2 = 128; s2 > 0; s2 >>= 1) { if (t < s2) ri[t] += ri[t + s2]; __syncthreads(); }
  if (t == 0) svalid = ri[0];
  __syncthreads();

  for (int r = 0; r < NSMP; ++r) {
    float lv = 2.f * BIGF; int li = 0;
    for (int i = t; i < NN; i += 256) {
      const float d = sd[i];
      if (d < lv) { lv = d; li = i; }
    }
    rv[t] = lv; ri[t] = li;
    __syncthreads();
    for (int s2 = 128; s2 > 0; s2 >>= 1) {
      if (t < s2) {
        if (rv[t + s2] < rv[t] || (rv[t + s2] == rv[t] && ri[t + s2] < ri[t])) {
          rv[t] = rv[t + s2]; ri[t] = ri[t + s2];
        }
      }
      __syncthreads();
    }
    if (t == 0) { sres[r] = ri[0]; sd[ri[0]] = 2.f * BIGF; }
    __syncthreads();
  }
  if (t < NSMP) {
    const int v = (svalid < NSMP) ? sres[0] : sres[t];
    knn_idx[((size_t)b * NPT + s) * NSMP + t] = v;
  }
}

// ---------------------------------------------------------------------------
// WMMA operand loaders (CDNA5 ISA 7.12.2 layouts).
// A (16x32 f16): lane 0-15 -> row M=lane, halves = K pairs {kh..kh+7, kh+16..};
//                lane 16-31 -> row M=lane-16, kh = 8.
// B (32x16 f16): lane -> column N=lane&15, 16 halves = K in A's order (kh = 8 for lanes 16-31).
//                Weights are pre-swizzled fragment-major in LDS so each lane's
//                operand is 32 contiguous bytes -> two ds_load_b128.
// D (16x16 f32): lane holds column N=lane&15, rows (lane>>4)*8 .. +7.
// ---------------------------------------------------------------------------
__device__ __forceinline__ v16h ldsA(const _Float16* __restrict__ m, int ldk,
                                     int m0, int kb, int lane) {
  const int row = m0 + (lane & 15);
  const int kh  = (lane >> 4) * 8;
  const _Float16* p = m + row * ldk + kb + kh;
  v8h lo = *(const v8h*)(p);        // K = kb+kh .. +7   (16B aligned)
  v8h hi = *(const v8h*)(p + 16);   // K = kb+16+kh .. +7
  return __builtin_shufflevector(lo, hi, 0, 1, 2, 3, 4, 5, 6, 7,
                                 8, 9, 10, 11, 12, 13, 14, 15);
}

// fragment-major weights: frag = ntile*NK + kchunk; element (frag, lane, k slot)
__device__ __forceinline__ v16h ldsBfrag(const _Float16* __restrict__ w,
                                         int frag, int lane) {
  return *(const v16h*)(w + (((size_t)frag * 32 + lane) << 4));
}

__device__ __forceinline__ void stD(_Float16* __restrict__ act, int ldn,
                                    int m0, int n0, v8f d,
                                    const float* __restrict__ bias, int lane) {
  const int n  = n0 + (lane & 15);
  const int mb = m0 + (lane >> 4) * 8;
  const float bb = bias[n];
  #pragma unroll
  for (int r = 0; r < 8; ++r) {
    const float x = d[r] + bb;
    act[(mb + r) * ldn + n] = (_Float16)(x > 0.f ? x : 0.f);
  }
}

// ---------------------------------------------------------------------------
// Kernel 3: fused gather + 3-layer MLP (v_wmma_f32_16x16x32_f16) + neighbor max.
// 8 waves cooperate per query; weights cached once (fragment-major f16) in LDS.
// ---------------------------------------------------------------------------
__global__ __launch_bounds__(256) void mlp_kernel(const float* __restrict__ xyz,
                                                  const float* __restrict__ points,
                                                  const int* __restrict__ fps_idx,
                                                  const int* __restrict__ knn_idx,
                                                  const float* __restrict__ w0,
                                                  const float* __restrict__ b0,
                                                  const float* __restrict__ w1,
                                                  const float* __restrict__ b1,
                                                  const float* __restrict__ w2,
                                                  const float* __restrict__ b2,
                                                  float* __restrict__ out) {
  // B-fragment-major weight caches: [frag][lane(32)][k slot(16)] halves
  __shared__ __attribute__((aligned(32))) _Float16 wg0[4 * 3 * 512];  // 67->pad96: NT=4, NK=3
  __shared__ __attribute__((aligned(32))) _Float16 wg1[4 * 2 * 512];  // NT=4, NK=2
  __shared__ __attribute__((aligned(32))) _Float16 wg2[8 * 2 * 512];  // NT=8, NK=2
  __shared__ float sb0[64], sb1[64], sb2[128];
  __shared__ __attribute__((aligned(16))) _Float16 feat[32 * 96];  // ns x Kpad
  __shared__ __attribute__((aligned(16))) _Float16 actA[32 * 64];
  __shared__ __attribute__((aligned(16))) _Float16 actB[32 * 64];
  const int t    = threadIdx.x;
  const int lane = t & 31;
  const int wv   = t >> 5;

  // swizzle weights into fragment-major layout:
  //   frag = nt*NK + kc ; k = kc*32 + lane2 ; o = nt*16 + n ; value = w[o][k]
  for (int e = t; e < 4 * 3 * 512; e += 256) {
    const int frag = e >> 9, r = e & 511, ln = r >> 4, s = r & 15, n = ln & 15, l2 = ((s >> 3) << 4) + ((ln >> 4) << 3) + (s & 7);
    const int nt = frag / 3, kc = frag % 3;
    const int c = kc * 32 + l2, o = nt * 16 + n;
    wg0[e] = (c < 67) ? (_Float16)w0[o * 67 + c] : (_Float16)0.f;
  }
  for (int e = t; e < 4 * 2 * 512; e += 256) {
    const int frag = e >> 9, r = e & 511, ln = r >> 4, s = r & 15, n = ln & 15, l2 = ((s >> 3) << 4) + ((ln >> 4) << 3) + (s & 7);
    const int nt = frag >> 1, kc = frag & 1;
    wg1[e] = (_Float16)w1[(nt * 16 + n) * 64 + kc * 32 + l2];
  }
  for (int e = t; e < 8 * 2 * 512; e += 256) {
    const int frag = e >> 9, r = e & 511, ln = r >> 4, s = r & 15, n = ln & 15, l2 = ((s >> 3) << 4) + ((ln >> 4) << 3) + (s & 7);
    const int nt = frag >> 1, kc = frag & 1;
    wg2[e] = (_Float16)w2[(nt * 16 + n) * 64 + kc * 32 + l2];
  }
  if (t < 64)  { sb0[t] = b0[t]; sb1[t] = b1[t]; }
  if (t < 128) sb2[t] = b2[t];
  __syncthreads();

  float* outp = out + (size_t)BB * NPT * 3;   // new_points after new_xyz

  for (int q = blockIdx.x; q < BB * NPT; q += gridDim.x) {
    const int b  = q >> 10;
    const int fi = fps_idx[q];
    const float* bx = xyz    + (size_t)b * NN * 3;
    const float* bp = points + (size_t)b * NN * CIN;
    const int* nidx = knn_idx + (size_t)q * NSMP;

    // gather: [xyz - center | point feats | zero pad] -> f16 LDS tile
    for (int e = t; e < 32 * 96; e += 256) {
      const int j = e / 96, c = e % 96;
      const int gi = nidx[j];
      float v;
      if (c < 3)       v = bx[gi * 3 + c] - bx[fi * 3 + c];
      else if (c < 67) v = bp[gi * CIN + (c - 3)];
      else             v = 0.f;
      feat[e] = (_Float16)v;
    }
    __syncthreads();

    { // layer 0: (32 x 96pad) x (96pad x 64) -> relu -> actA
      const int mi = wv & 1, ni = wv >> 1;
      v8f acc = {};
      #pragma unroll
      for (int ks = 0; ks < 3; ++ks) {
        v16h A  = ldsA(feat, 96, mi * 16, ks * 32, lane);
        v16h Bv = ldsBfrag(wg0, ni * 3 + ks, lane);
        acc = __builtin_amdgcn_wmma_f32_16x16x32_f16(false, A, false, Bv,
                                                     (short)0, acc, false, false);
      }
      stD(actA, 64, mi * 16, ni * 16, acc, sb0, lane);
    }
    __syncthreads();

    { // layer 1: (32 x 64) x (64 x 64) -> relu -> actB
      const int mi = wv & 1, ni = wv >> 1;
      v8f acc = {};
      #pragma unroll
      for (int ks = 0; ks < 2; ++ks) {
        v16h A  = ldsA(actA, 64, mi * 16, ks * 32, lane);
        v16h Bv = ldsBfrag(wg1, ni * 2 + ks, lane);
        acc = __builtin_amdgcn_wmma_f32_16x16x32_f16(false, A, false, Bv,
                                                     (short)0, acc, false, false);
      }
      stD(actB, 64, mi * 16, ni * 16, acc, sb1, lane);
    }
    __syncthreads();

    { // layer 2: (32 x 64) x (64 x 128), then max over the 32 neighbors
      v8f a0 = {}, a1 = {};
      #pragma unroll
      for (int ks = 0; ks < 2; ++ks) {
        v16h Bv = ldsBfrag(wg2, wv * 2 + ks, lane);
        v16h A0 = ldsA(actB, 64, 0,  ks * 32, lane);
        v16h A1 = ldsA(actB, 64, 16, ks * 32, lane);
        a0 = __builtin_amdgcn_wmma_f32_16x16x32_f16(false, A0, false, Bv,
                                                    (short)0, a0, false, false);
        a1 = __builtin_amdgcn_wmma_f32_16x16x32_f16(false, A1, false, Bv,
                                                    (short)0, a1, false, false);
      }
      float mx = -3.0e38f;
      #pragma unroll
      for (int r = 0; r < 8; ++r) { mx = fmaxf(mx, a0[r]); mx = fmaxf(mx, a1[r]); }
      mx = fmaxf(mx, __shfl_xor(mx, 16, 32));   // combine row halves (same column)
      if (lane < 16) {
        const int col = wv * 16 + lane;
        // max(relu(x+b)) == relu(max(x)+b): bias is per-column constant
        outp[(size_t)q * 128 + col] = fmaxf(mx + sb2[col], 0.f);
      }
    }
    __syncthreads();   // protect feat/act reuse across queries
  }
}

// ---------------------------------------------------------------------------
extern "C" void kernel_launch(void* const* d_in, const int* in_sizes, int n_in,
                              void* d_out, int out_size, void* d_ws, size_t ws_size,
                              hipStream_t stream) {
  (void)in_sizes; (void)n_in; (void)out_size; (void)ws_size;
  const float* xyz    = (const float*)d_in[0];
  const float* points = (const float*)d_in[1];
  const int*   mask   = (const int*)d_in[2];
  const float* w0 = (const float*)d_in[3];
  const float* b0 = (const float*)d_in[4];
  const float* w1 = (const float*)d_in[5];
  const float* b1 = (const float*)d_in[6];
  const float* w2 = (const float*)d_in[7];
  const float* b2 = (const float*)d_in[8];
  float* out = (float*)d_out;

  int* fps_idx = (int*)d_ws;                         // B*NPT ints
  int* knn_idx = fps_idx + (size_t)BB * NPT;         // B*NPT*NSMP ints

  fps_kernel<<<BB, 256, 0, stream>>>(xyz, mask, fps_idx);
  knn_kernel<<<BB * NPT, 256, 0, stream>>>(xyz, mask, fps_idx, knn_idx, out);
  mlp_kernel<<<2048, 256, 0, stream>>>(xyz, points, fps_idx, knn_idx,
                                       w0, b0, w1, b1, w2, b2, out);
}
